// MultiheadSelectiveAttention_10179072491724
// MI455X (gfx1250) — matrix operand order repaired and checked by reading, not compiled, not run on hardware
//
#include <hip/hip_runtime.h>

// MI455X / gfx1250, wave32. Compute-bound (~206 GFLOP vs ~200MB @ 23.3 TB/s):
// all matmuls on v_wmma_f32_16x16x32_bf16, fed by async global->LDS DMA
// (global_load_async_to_lds_b128, ASYNCcnt) with double-buffered LDS tiles.

#define D_MODEL 2048
#define NHEADS  16
#define DHEAD   128
#define SEQ     2048
#define BATCH   2
#define ROWS    (BATCH * SEQ)   // 4096

typedef __attribute__((ext_vector_type(16))) __bf16        v16bf;
typedef __attribute__((ext_vector_type(8)))  float         v8f;
typedef __attribute__((ext_vector_type(8)))  unsigned int  v8u;

union Frag16 { v16bf bf; v8u u; };

__device__ inline unsigned short f32_to_bf16(float f) {
  unsigned int u = __float_as_uint(f);
  u += 0x7FFFu + ((u >> 16) & 1u);          // round-to-nearest-even
  return (unsigned short)(u >> 16);
}

__device__ inline v8f wmma_bf16(const Frag16& a, const Frag16& b, v8f c) {
  return __builtin_amdgcn_wmma_f32_16x16x32_bf16(false, a.bf, false, b.bf,
                                                 (short)0, c, false, false);
}

// ---- async global -> LDS (CDNA5 TDM-lite path, tracked by ASYNCcnt) ----
__device__ inline unsigned int lds_off(const void* p) {
  // generic address of a __shared__ object: low 32 bits are the LDS offset
  return (unsigned int)(unsigned long long)(size_t)p;
}
__device__ inline void async_ld_b128(unsigned int ldsAddr, const void* g) {
  asm volatile("global_load_async_to_lds_b128 %0, %1, off"
               :: "v"(ldsAddr), "v"((unsigned long long)(size_t)g) : "memory");
}
__device__ inline void wait_async0() {
  asm volatile("s_wait_asynccnt 0x0" ::: "memory");
}

// ---- fragment loaders ----
// Contiguous 32B per lane: used for the swizzled A tiles in
// LDS. p must point at (rowOrCol base + (lane>>4)*16) elements.
__device__ inline void load_frag32(Frag16& f, const unsigned short* p) {
  const uint4 q0 = *(const uint4*)p;
  const uint4 q1 = *(const uint4*)(p + 8);
  f.u[0] = q0.x; f.u[1] = q0.y; f.u[2] = q0.z; f.u[3] = q0.w;
  f.u[4] = q1.x; f.u[5] = q1.y; f.u[6] = q1.z; f.u[7] = q1.w;
}
// A fragment from a natural (unswizzled) K-contiguous row: two 16B runs.
__device__ inline void load_a_frag_nat(Frag16& f, const unsigned short* rowPtr, int lane) {
  const int g8 = (lane >> 4) * 8;
  const uint4 q0 = *(const uint4*)(rowPtr + g8);
  const uint4 q1 = *(const uint4*)(rowPtr + 16 + g8);
  f.u[0] = q0.x; f.u[1] = q0.y; f.u[2] = q0.z; f.u[3] = q0.w;
  f.u[4] = q1.x; f.u[5] = q1.y; f.u[6] = q1.z; f.u[7] = q1.w;
}
// B fragment (natural): pairs (lane>>4)*8 .. +7 are contiguous.
__device__ inline void load_b_frag_nat(Frag16& f, const unsigned short* colPtr, int lane) {
  const int g16 = (lane >> 4) * 16;
  const uint4 q0 = *(const uint4*)(colPtr + g16);
  const uint4 q1 = *(const uint4*)(colPtr + g16 + 8);
  f.u[0] = q0.x; f.u[1] = q0.y; f.u[2] = q0.z; f.u[3] = q0.w;
  f.u[4] = q1.x; f.u[5] = q1.y; f.u[6] = q1.z; f.u[7] = q1.w;
}
// A fragment from scattered (natural) LDS row — only used for the tiny P bounce.
__device__ inline void load_a_frag_scat(Frag16& f, const unsigned short* rowPtr, int lane) {
  const unsigned int* p = (const unsigned int*)rowPtr;
  const int kg = (lane >> 4) * 4;
#pragma unroll
  for (int v = 0; v < 4; ++v) f.u[v] = p[kg + v];
#pragma unroll
  for (int v = 0; v < 4; ++v) f.u[4 + v] = p[8 + kg + v];
}

// ---------------------------------------------------------------- convert
__global__ __launch_bounds__(256)
void cvt_f32_bf16(const float* __restrict__ in, unsigned short* __restrict__ out, long n) {
  const long n4 = n >> 2;
  const float4* in4 = (const float4*)in;
  uint2* out2 = (uint2*)out;
  long i = (long)blockIdx.x * blockDim.x + threadIdx.x;
  const long stride = (long)gridDim.x * blockDim.x;
  for (; i < n4; i += stride) {
    const float4 f = in4[i];
    uint2 r;
    r.x = ((unsigned int)f32_to_bf16(f.y) << 16) | f32_to_bf16(f.x);
    r.y = ((unsigned int)f32_to_bf16(f.w) << 16) | f32_to_bf16(f.z);
    out2[i] = r;
  }
}

// ---------------------------------------------------------------- NT GEMM
// C[M,N] = A[M,K] @ B[N,K]^T, bf16 in / f32 out. 128x128 tile, 8 waves,
// K chunks of 32, double-buffered async global->LDS. A tile stored swizzled
// (row pair order [0-3, 8-11, 4-7, 12-15]) so every fragment is one
// contiguous 32B per-lane read.
__global__ __launch_bounds__(256)
void gemm_nt_bf16(const unsigned short* __restrict__ A,
                  const unsigned short* __restrict__ B,
                  float* __restrict__ C, int M, int N, int K) {
  __shared__ __align__(16) unsigned short lA[2][128 * 32];
  __shared__ __align__(16) unsigned short lB[2][128 * 32];
  const int tid  = threadIdx.x;
  const int lane = tid & 31;
  const int wave = tid >> 5;
  const int wm = wave >> 1;                 // 0..3 -> 32-row strip
  const int wn = wave & 1;                  // 0..1 -> 64-col strip
  const int mBase = blockIdx.y * 128;
  const int nBase = blockIdx.x * 128;
  const int c16 = lane & 15;
  const int g16 = (lane >> 4) * 16;

  v8f acc[2][4] = {};

  const int ldRow = tid >> 1;               // 0..127
  const int seg   = tid & 1;                // which 16-element half of chunk
  const unsigned short* gA = A + (size_t)(mBase + ldRow) * K + seg * 16;
  const unsigned short* gB = B + (size_t)(nBase + ldRow) * K + seg * 16;
  // swizzled A destinations; natural B destinations
  const int dA0 = ldRow * 32 + (seg ? 8 : 0);
  const int dA1 = ldRow * 32 + (seg ? 24 : 16);
  const int dB0 = ldRow * 32 + seg * 16;

  // prologue: async-fill buffer 0 with K-chunk 0
  async_ld_b128(lds_off(&lA[0][dA0]), gA);
  async_ld_b128(lds_off(&lA[0][dA1]), gA + 8);
  async_ld_b128(lds_off(&lB[0][dB0]), gB);
  async_ld_b128(lds_off(&lB[0][dB0 + 8]), gB + 8);
  wait_async0();
  __syncthreads();

  for (int kb = 0; kb < K; kb += 32) {
    const int cur = (kb >> 5) & 1;
    const int nxt = cur ^ 1;
    if (kb + 32 < K) {                      // overlap next chunk's DMA
      async_ld_b128(lds_off(&lA[nxt][dA0]), gA + kb + 32);
      async_ld_b128(lds_off(&lA[nxt][dA1]), gA + kb + 40);
      async_ld_b128(lds_off(&lB[nxt][dB0]), gB + kb + 32);
      async_ld_b128(lds_off(&lB[nxt][dB0 + 8]), gB + kb + 40);
    }

    Frag16 af[2], bf4[4];
#pragma unroll
    for (int mf = 0; mf < 2; ++mf)
      load_frag32(af[mf], &lA[cur][(wm * 32 + mf * 16 + c16) * 32 + g16]);
#pragma unroll
    for (int nf = 0; nf < 4; ++nf)
      load_a_frag_nat(bf4[nf], &lB[cur][(wn * 64 + nf * 16 + c16) * 32], lane);
#pragma unroll
    for (int mf = 0; mf < 2; ++mf)
#pragma unroll
      for (int nf = 0; nf < 4; ++nf)
        acc[mf][nf] = wmma_bf16(af[mf], bf4[nf], acc[mf][nf]);

    wait_async0();                          // own DMA landed
    __syncthreads();                        // everyone's DMA landed / reads done
  }

  const int rh = (lane >> 4) * 8;
#pragma unroll
  for (int mf = 0; mf < 2; ++mf)
#pragma unroll
    for (int nf = 0; nf < 4; ++nf)
#pragma unroll
      for (int v = 0; v < 8; ++v) {
        const int gr = mBase + wm * 32 + mf * 16 + rh + v;
        const int gc = nBase + wn * 64 + nf * 16 + c16;
        C[(size_t)gr * N + gc] = acc[mf][nf][v];
      }
}

// ---------------------------------------------------------------- LayerNorm
__global__ __launch_bounds__(256)
void layernorm_bf16(const float* __restrict__ in, const float* __restrict__ gamma,
                    const float* __restrict__ beta, unsigned short* __restrict__ out) {
  const int row = blockIdx.x;
  const float* x = in + (size_t)row * D_MODEL;
  float s = 0.f, s2 = 0.f;
  for (int i = threadIdx.x; i < D_MODEL; i += 256) {
    const float v = x[i]; s += v; s2 += v * v;
  }
#pragma unroll
  for (int m = 16; m >= 1; m >>= 1) {
    s  += __shfl_xor(s, m, 32);
    s2 += __shfl_xor(s2, m, 32);
  }
  __shared__ float red[16];
  if ((threadIdx.x & 31) == 0) {
    red[threadIdx.x >> 5]       = s;
    red[8 + (threadIdx.x >> 5)] = s2;
  }
  __syncthreads();
  float S = 0.f, S2 = 0.f;
#pragma unroll
  for (int w2 = 0; w2 < 8; ++w2) { S += red[w2]; S2 += red[8 + w2]; }
  const float mean = S * (1.f / D_MODEL);
  const float var  = S2 * (1.f / D_MODEL) - mean * mean;
  const float rstd = rsqrtf(var + 1e-5f);
  unsigned short* o = out + (size_t)row * D_MODEL;
  for (int i = threadIdx.x; i < D_MODEL; i += 256)
    o[i] = f32_to_bf16((x[i] - mean) * rstd * gamma[i] + beta[i]);
}

// ---------------------------------------------------------------- head-0 S
__global__ __launch_bounds__(256)
void head0_s_kernel(const unsigned short* __restrict__ Q,
                    const unsigned short* __restrict__ K,
                    float* __restrict__ F) {
  const int lane = threadIdx.x & 31;
  const int wave = threadIdx.x >> 5;
  const long t   = (long)blockIdx.x * 8 + wave;
  const int b   = (int)(t / (128 * 128));
  const int rem = (int)(t % (128 * 128));
  const int tr = rem / 128;
  const int tc = rem % 128;
  const float scale = 0.088388347648318447f;   // 1/sqrt(128)

  v8f acc = {};
  if (tc <= tr) {
    const unsigned short* qrow = Q + (size_t)(b * SEQ + tr * 16 + (lane & 15)) * D_MODEL;
    const unsigned short* krow = K + (size_t)(b * SEQ + tc * 16 + (lane & 15)) * D_MODEL;
#pragma unroll
    for (int kb = 0; kb < DHEAD; kb += 32) {
      Frag16 a, bb;
      load_a_frag_nat(a, qrow + kb, lane);
      load_a_frag_nat(bb, krow + kb, lane);
      acc = wmma_bf16(a, bb, acc);
    }
  }
  const int col = tc * 16 + (lane & 15);
  const int rh  = (lane >> 4) * 8;
#pragma unroll
  for (int v = 0; v < 8; ++v) {
    const int i = tr * 16 + rh + v;
    float val = fmaxf(acc[v] * scale, 0.f);
    if (col >= i || col == 0) val = 0.f;     // causal+diag+col0 zeros
    if (i + 1 < SEQ)
      F[((size_t)b * SEQ + (i + 1)) * SEQ + col] = val;   // roll down 1 row
  }
  if (tr == 0 && lane < 16)
    F[(size_t)b * SEQ * SEQ + col] = 0.f;    // row 0 of shifted S
}

// ---------------------------------------------------------------- cumsum rows
__global__ __launch_bounds__(256)
void cumsum_rows(float* __restrict__ F) {
  const int b   = blockIdx.x / (SEQ / 256);
  const int col = (blockIdx.x % (SEQ / 256)) * 256 + threadIdx.x;
  float* base = F + (size_t)b * SEQ * SEQ + col;
  float acc = 0.f;
  for (int r = 0; r < SEQ; r += 8) {
    float v[8];
#pragma unroll
    for (int j = 0; j < 8; ++j) v[j] = base[(size_t)(r + j) * SEQ];
#pragma unroll
    for (int j = 0; j < 8; ++j) { acc += v[j]; base[(size_t)(r + j) * SEQ] = acc; }
  }
}

// ---------------------------------------------------------------- attention
// Flash-style: 4 waves x 16 query rows, key tiles of 32, QK^T and P.V on
// WMMA. Q tile stored swizzled for contiguous A fragments; K tile filled by
// async DMA; V transposed manually for B fragments.
__global__ __launch_bounds__(128)
void attention_kernel(const unsigned short* __restrict__ Q,
                      const unsigned short* __restrict__ K,
                      const unsigned short* __restrict__ V,
                      const float* __restrict__ F,
                      unsigned short* __restrict__ O) {
  __shared__ __align__(16) unsigned short lq[64 * 128];    // Q tile, swizzled chunks
  __shared__ __align__(16) unsigned short lk[32 * 128];    // K tile, row-major
  __shared__ __align__(16) unsigned short lvt[128 * 32];   // V tile, [dh][key]
  __shared__ __align__(16) unsigned short lp[4 * 16 * 32]; // per-wave P bounce

  const int tid  = threadIdx.x;
  const int lane = tid & 31;
  const int w    = tid >> 5;                 // 0..3
  const int qt = blockIdx.x & 31;
  const int h  = (blockIdx.x >> 5) & 15;
  const int b  = blockIdx.x >> 9;
  const float scale = 0.088388347648318447f;
  const float NEG = -1e30f;
  const int c16 = lane & 15;
  const int rh  = (lane >> 4) * 8;
  const int g16 = (lane >> 4) * 16;

  { // load Q tile 64x128, swizzling each 32-element K chunk for A fragments
    const int r = tid >> 1;
    const int half = (tid & 1) * 64;
    const uint4* g = (const uint4*)(Q + (size_t)(b * SEQ + qt * 64 + r) * D_MODEL
                                      + h * DHEAD + half);
    const int map[4] = {0, 16, 8, 24};
#pragma unroll
    for (int j = 0; j < 8; ++j) {
      const int chunk = j >> 2, jj = j & 3;
      *(uint4*)&lq[r * 128 + half + chunk * 32 + map[jj]] = g[j];
    }
  }

  v8f oac[8] = {};
  float mst[8], lst[8];
#pragma unroll
  for (int v = 0; v < 8; ++v) { mst[v] = NEG; lst[v] = 0.f; }

  const int nTiles = qt * 2 + 2;             // causal bound (key <= row)

  for (int kt = 0; kt < nTiles; ++kt) {
    const int keyBase = kt * 32;
    __syncthreads();                         // prev tile fully consumed
    { // K tile via async DMA; V tile transposed manually
      const int r  = tid >> 2;               // 0..31
      const int qo = (tid & 3) * 32;
      const unsigned short* gk = K + (size_t)(b * SEQ + keyBase + r) * D_MODEL
                                   + h * DHEAD + qo;
#pragma unroll
      for (int j = 0; j < 4; ++j)
        async_ld_b128(lds_off(&lk[r * 128 + qo + j * 8]), gk + j * 8);

      uint4 tv[4];
      const uint4* gv = (const uint4*)(V + (size_t)(b * SEQ + keyBase + r) * D_MODEL
                                         + h * DHEAD + qo);
#pragma unroll
      for (int j = 0; j < 4; ++j) tv[j] = gv[j];
      const unsigned short* ts = (const unsigned short*)tv;
#pragma unroll
      for (int e = 0; e < 32; ++e)
        lvt[(qo + e) * 32 + r] = ts[e];
    }
    wait_async0();
    __syncthreads();

    // S = Q K^T (16 rows x 32 keys per wave), K=128 in 4 chunks
    v8f sac[2] = {};
#pragma unroll
    for (int kb = 0; kb < 4; ++kb) {
      Frag16 a;
      load_frag32(a, lq + (w * 16 + c16) * 128 + kb * 32 + g16);
#pragma unroll
      for (int nf = 0; nf < 2; ++nf) {
        Frag16 bb;
        load_a_frag_nat(bb, lk + (nf * 16 + c16) * 128 + kb * 32, lane);
        sac[nf] = wmma_bf16(a, bb, sac[nf]);
      }
    }

    // masked logits: scale, subtract F_mask, causal
    float sv[2][8];
#pragma unroll
    for (int nf = 0; nf < 2; ++nf) {
      const int key = keyBase + nf * 16 + c16;
#pragma unroll
      for (int v = 0; v < 8; ++v) {
        const int row = qt * 64 + w * 16 + rh + v;
        float x = sac[nf][v] * scale - F[((size_t)b * SEQ + row) * SEQ + key];
        if (key > row) x = NEG;
        sv[nf][v] = x;
      }
    }

    // online softmax; row lives in one VGPR across 16 lanes -> xor masks 1..8
    float alpha[8], mnew[8];
#pragma unroll
    for (int v = 0; v < 8; ++v) {
      float mx = fmaxf(sv[0][v], sv[1][v]);
#pragma unroll
      for (int m = 8; m >= 1; m >>= 1) mx = fmaxf(mx, __shfl_xor(mx, m, 32));
      mnew[v]  = fmaxf(mst[v], mx);
      alpha[v] = __expf(mst[v] - mnew[v]);
      mst[v]   = mnew[v];
    }
#pragma unroll
    for (int v = 0; v < 8; ++v) {
      const float p0 = __expf(sv[0][v] - mnew[v]);
      const float p1 = __expf(sv[1][v] - mnew[v]);
      sv[0][v] = p0; sv[1][v] = p1;
      float rs = p0 + p1;
#pragma unroll
      for (int m = 8; m >= 1; m >>= 1) rs += __shfl_xor(rs, m, 32);
      lst[v] = lst[v] * alpha[v] + rs;
    }
#pragma unroll
    for (int d = 0; d < 8; ++d)
#pragma unroll
      for (int v = 0; v < 8; ++v)
        oac[d][v] *= alpha[v];

    // P (C-layout) -> per-wave LDS (natural 16x32) -> A fragment
#pragma unroll
    for (int nf = 0; nf < 2; ++nf)
#pragma unroll
      for (int v = 0; v < 8; ++v)
        lp[w * 512 + (rh + v) * 32 + nf * 16 + c16] = f32_to_bf16(sv[nf][v]);

    Frag16 ap;
    load_a_frag_scat(ap, lp + w * 512 + c16 * 32, lane);
#pragma unroll
    for (int d = 0; d < 8; ++d) {
      Frag16 bv;
      load_a_frag_nat(bv, lvt + (d * 16 + c16) * 32, lane);
      oac[d] = wmma_bf16(ap, bv, oac[d]);
    }
  }

  // O = acc / l, bf16, [B*N, D] with head offset
#pragma unroll
  for (int d = 0; d < 8; ++d)
#pragma unroll
    for (int v = 0; v < 8; ++v) {
      const int row = b * SEQ + qt * 64 + w * 16 + rh + v;
      const int col = h * DHEAD + d * 16 + c16;
      O[(size_t)row * D_MODEL + col] = f32_to_bf16(oac[d][v] / lst[v]);
    }
}

// ---------------------------------------------------------------- launch
extern "C" void kernel_launch(void* const* d_in, const int* in_sizes, int n_in,
                              void* d_out, int out_size, void* d_ws, size_t ws_size,
                              hipStream_t stream) {
  (void)in_sizes; (void)n_in; (void)out_size; (void)ws_size;
  const float* X  = (const float*)d_in[0];
  const float* Wq = (const float*)d_in[1];
  const float* Wk = (const float*)d_in[2];
  const float* Wv = (const float*)d_in[3];
  const float* Wo = (const float*)d_in[4];
  const float* gq = (const float*)d_in[5];
  const float* bq = (const float*)d_in[6];
  const float* gk = (const float*)d_in[7];
  const float* bk = (const float*)d_in[8];

  float* out = (float*)d_out;
  float* Fm  = out + (size_t)BATCH * SEQ * D_MODEL;   // F_mask region

  char* ws = (char*)d_ws;
  const size_t MB = 1024 * 1024;
  unsigned short* Xbf = (unsigned short*)(ws);            // 16MB (reused as Obf)
  unsigned short* Wbf = (unsigned short*)(ws + 16 * MB);  //  8MB (one weight at a time)
  float*          tmp = (float*)(ws + 24 * MB);           // 32MB f32 GEMM output
  unsigned short* Qbf = (unsigned short*)(ws + 56 * MB);  // 16MB
  unsigned short* Kbf = (unsigned short*)(ws + 72 * MB);  // 16MB
  unsigned short* Vbf = (unsigned short*)(ws + 88 * MB);  // 16MB   (104MB total)
  unsigned short* Obf = Xbf;

  const long nX = (long)ROWS * D_MODEL;
  const long nW = (long)D_MODEL * D_MODEL;
  dim3 gg(D_MODEL / 128, ROWS / 128);                     // (16, 32)

  cvt_f32_bf16<<<2048, 256, 0, stream>>>(X, Xbf, nX);

  // Q = LN(X Wq^T)
  cvt_f32_bf16<<<1024, 256, 0, stream>>>(Wq, Wbf, nW);
  gemm_nt_bf16<<<gg, 256, 0, stream>>>(Xbf, Wbf, tmp, ROWS, D_MODEL, D_MODEL);
  layernorm_bf16<<<ROWS, 256, 0, stream>>>(tmp, gq, bq, Qbf);
  // K = LN(X Wk^T)
  cvt_f32_bf16<<<1024, 256, 0, stream>>>(Wk, Wbf, nW);
  gemm_nt_bf16<<<gg, 256, 0, stream>>>(Xbf, Wbf, tmp, ROWS, D_MODEL, D_MODEL);
  layernorm_bf16<<<ROWS, 256, 0, stream>>>(tmp, gk, bk, Kbf);
  // V = X Wv^T
  cvt_f32_bf16<<<1024, 256, 0, stream>>>(Wv, Wbf, nW);
  gemm_nt_bf16<<<gg, 256, 0, stream>>>(Xbf, Wbf, tmp, ROWS, D_MODEL, D_MODEL);
  cvt_f32_bf16<<<2048, 256, 0, stream>>>(tmp, Vbf, nX);

  // selective forgetting mask
  head0_s_kernel<<<(BATCH * 128 * 128) / 8, 256, 0, stream>>>(Qbf, Kbf, Fm);
  cumsum_rows<<<BATCH * (SEQ / 256), 256, 0, stream>>>(Fm);

  // attention (Xbf no longer needed -> reuse as O)
  attention_kernel<<<BATCH * NHEADS * (SEQ / 64), 128, 0, stream>>>(Qbf, Kbf, Vbf, Fm, Obf);

  // output projection -> d_out
  cvt_f32_bf16<<<1024, 256, 0, stream>>>(Wo, Wbf, nW);
  gemm_nt_bf16<<<gg, 256, 0, stream>>>(Obf, Wbf, out, ROWS, D_MODEL, D_MODEL);
}
